// CrossAttention_77249281786046
// MI455X (gfx1250) — matrix operand order repaired and checked by reading, not compiled, not run on hardware
//
#include <hip/hip_runtime.h>
#include <hip/hip_bf16.h>

#define DIMX    256
#define NHEADS  8
#define HDIM    32
#define BATCH   2
#define NQ      8192            // query rows per batch (power of two)
#define MKV     2048            // key rows per batch (power of two)
#define QSCALE  0.17677669529663687f   // 1/sqrt(32)
#define KBLK    8               // 256 / 32 K-blocks per GEMM

typedef __attribute__((ext_vector_type(16))) _Float16 v16h;
typedef __attribute__((ext_vector_type(8)))  _Float16 v8h;
typedef __attribute__((ext_vector_type(8)))  float    v8f;

// ---------------------------------------------------------------------------
// WMMA helper: D = A(16x32 f16) * B(32x16 f16) + C(16x16 f32)
// ---------------------------------------------------------------------------
__device__ __forceinline__ v8f wmma_f16(v16h a, v16h b, v8f c) {
    return __builtin_amdgcn_wmma_f32_16x16x32_f16(
        /*neg_a=*/false, a, /*neg_b=*/false, b,
        /*c_mod=*/(short)0, c, /*reuse_a=*/false, /*reuse_b=*/false);
}

// A-fragment (16x32, f16), contiguous k order used by this file (B matches):
//   lanes 0-15 : row = lane,    halves 0-15 -> K=0..15
//   lanes 16-31: row = lane-16, halves 0-15 -> K=16..31
__device__ __forceinline__ v16h frag_a_from_h(const _Float16* src, int ld, int lane) {
    const int r  = lane & 15;
    const int kb = (lane & 16);               // 0 or 16
    const _Float16* p = src + r * ld + kb;
    v8h lo = *(const v8h*)(p);                // K = kb .. kb+7
    v8h hi = *(const v8h*)(p + 8);            // K = kb+8 .. kb+15
    v16h a;
#pragma unroll
    for (int i = 0; i < 8; ++i) { a[i] = lo[i]; a[8 + i] = hi[i]; }
    return a;
}

// ---------------------------------------------------------------------------
// One-time fp32 -> f16 elementwise convert (8 elements / thread, b128 in/out)
// ---------------------------------------------------------------------------
__global__ void cvt_f16_kernel(const float* __restrict__ src,
                               _Float16* __restrict__ dst) {
    const size_t idx = (size_t)blockIdx.x * blockDim.x + threadIdx.x;
    v8f s = *(const v8f*)(src + idx * 8);
    v8h d;
#pragma unroll
    for (int i = 0; i < 8; ++i) d[i] = (_Float16)s[i];
    *(v8h*)(dst + idx * 8) = d;
}

// ---------------------------------------------------------------------------
// One-time weight pack: W (256 x outc fp32, K-major) -> fragment-ready f16
// layout [colTile][kBlk][lane][16], so a B-fragment is ONE contiguous 32B load.
// B-fragment layout used here (contiguous k): lanes 0-15: col=lane, K=0..15; lanes 16-31: K=16..31.
// grid = (outc/16, 8), block = 32
// ---------------------------------------------------------------------------
__global__ void pack_w_kernel(const float* __restrict__ W,
                              _Float16* __restrict__ Wp, int ldw) {
    const int lane = threadIdx.x;
    const int t    = blockIdx.x;                    // column tile
    const int kk   = blockIdx.y;                    // K block of 32
    const int c    = t * 16 + (lane & 15);
    const int K0   = kk * 32 + (lane & 16);
    v16h b;
#pragma unroll
    for (int i = 0; i < 16; ++i) b[i] = (_Float16)W[(size_t)(K0 + i) * ldw + c];
    *(v16h*)(Wp + (((size_t)t * KBLK + kk) * 32 + lane) * 16) = b;
}

// ---------------------------------------------------------------------------
// Core GEMM tile: 16 rows x (4*16) cols, K=256. A loaded once per K-block and
// reused across 4 back-to-back WMMAs (packed B = contiguous 32B loads).
// ---------------------------------------------------------------------------
__device__ __forceinline__ void gemm16x64(const _Float16* __restrict__ Arow,
                                          const _Float16* __restrict__ Wp,
                                          int t0, int lane, v8f acc[4]) {
#pragma unroll
    for (int kk = 0; kk < KBLK; ++kk) {
        const v16h a = frag_a_from_h(Arow + kk * 32, DIMX, lane);
#pragma unroll
        for (int t = 0; t < 4; ++t) {
            const v16h b = *(const v16h*)(Wp + (((size_t)(t0 + t) * KBLK + kk) * 32 + lane) * 16);
            acc[t] = wmma_f16(a, b, acc[t]);
        }
    }
}

// ---------------------------------------------------------------------------
// Q projection: qf[bh][n][hd] = f16( x @ Wq + bq )
// grid = (B*N/16, 4), block = 32 (one wave -> 16x64 outputs)
// ---------------------------------------------------------------------------
__global__ void q_proj_kernel(const _Float16* __restrict__ xh,
                              const _Float16* __restrict__ WqP,
                              const float* __restrict__ bq,
                              _Float16* __restrict__ qf) {
    const int lane = threadIdx.x;
    const int row0 = blockIdx.x * 16;              // row in B*N
    const int t0   = blockIdx.y * 4;               // first col tile
    v8f acc[4] = {};
    gemm16x64(xh + (size_t)row0 * DIMX, WqP, t0, lane, acc);

    const int n   = lane & 15;
    const int mrb = (lane & 16) >> 1;
#pragma unroll
    for (int t = 0; t < 4; ++t) {
        const int gcol = (t0 + t) * 16 + n;
        const float bias = bq[gcol];
        const int h = gcol >> 5, d = gcol & 31;
#pragma unroll
        for (int i = 0; i < 8; ++i) {
            const int grow = row0 + mrb + i;                    // in B*N
            const int b_ = grow >> 13, n_ = grow & (NQ - 1);
            qf[((size_t)(b_ * NHEADS + h) * NQ + n_) * HDIM + d] =
                (_Float16)(acc[t][i] + bias);
        }
    }
}

// ---------------------------------------------------------------------------
// KV projection: kf[bh][m][hd] row-major, vT[bh][hd][m] transposed.
// grid = (B*M/16, 8), block = 32
// ---------------------------------------------------------------------------
__global__ void kv_proj_kernel(const _Float16* __restrict__ ch,
                               const _Float16* __restrict__ WkvP,
                               const float* __restrict__ bkv,
                               _Float16* __restrict__ kf,
                               _Float16* __restrict__ vT) {
    const int lane = threadIdx.x;
    const int row0 = blockIdx.x * 16;              // row in B*M
    const int t0   = blockIdx.y * 4;
    v8f acc[4] = {};
    gemm16x64(ch + (size_t)row0 * DIMX, WkvP, t0, lane, acc);

    const int n   = lane & 15;
    const int mrb = (lane & 16) >> 1;
#pragma unroll
    for (int t = 0; t < 4; ++t) {
        const int gcol = (t0 + t) * 16 + n;
        const float bias = bkv[gcol];
        const int kv_sel = gcol >> 8;              // 0 = K, 1 = V
        const int h = (gcol >> 5) & 7, d = gcol & 31;
#pragma unroll
        for (int i = 0; i < 8; ++i) {
            const int grow = row0 + mrb + i;                    // in B*M
            const int b_ = grow >> 11, m_ = grow & (MKV - 1);
            const float val = acc[t][i] + bias;
            if (kv_sel == 0)
                kf[((size_t)(b_ * NHEADS + h) * MKV + m_) * HDIM + d] = (_Float16)val;
            else
                vT[((size_t)(b_ * NHEADS + h) * HDIM + d) * MKV + m_] = (_Float16)val;
        }
    }
}

// ---------------------------------------------------------------------------
// Flash attention core: one wave = 16 query rows of one (b,h).
// Online softmax; S and P never leave the WGP. Output written as f16 (B,N,256).
// ---------------------------------------------------------------------------
__global__ void flash_attn_kernel(const _Float16* __restrict__ qf,
                                  const _Float16* __restrict__ kf,
                                  const _Float16* __restrict__ vT,
                                  const unsigned char* __restrict__ mask,
                                  _Float16* __restrict__ aoh) {
    __shared__ __attribute__((aligned(32))) _Float16 ldsP[16 * 32];  // P tile

    const int lane = threadIdx.x;
    const int bh   = blockIdx.x;          // B*H
    const int q0   = blockIdx.y * 16;
    const int b    = bh >> 3;
    const int h    = bh & 7;

    const _Float16* qbase = qf + ((size_t)bh * NQ + q0) * HDIM;
    const _Float16* kbase = kf + (size_t)bh * MKV * HDIM;
    const _Float16* vbase = vT + (size_t)bh * HDIM * MKV;

    const v16h aQ = frag_a_from_h(qbase, HDIM, lane);

    const int n    = lane & 15;           // column within 16-wide tile
    const int K0   = lane & 16;           // B-fragment K offset (0 or 16)
    const int rowb = (lane & 16) >> 1;    // C/D rows owned: rowb .. rowb+7

    float rowOK[8], mI[8], lI[8];
    v8f o0 = {}, o1 = {};
#pragma unroll
    for (int i = 0; i < 8; ++i) {
        rowOK[i] = mask[(size_t)b * NQ + q0 + rowb + i] ? 1.0f : 0.0f;
        mI[i] = -3.0e38f;
        lI[i] = 0.0f;
    }

    for (int j = 0; j < MKV; j += 32) {
        if (j + 32 < MKV) {   // global_prefetch next key/value block
            __builtin_prefetch(kbase + (size_t)(j + 32 + n) * HDIM, 0, 1);
            __builtin_prefetch(vbase + (size_t)n * MKV + j + 32, 0, 1);
        }
        // ---- S = Q @ K^T for keys j..j+31 (B-frag = contiguous 32B/lane)
        v16h bK0 = *(const v16h*)(kbase + (size_t)(j + n) * HDIM + K0);
        v16h bK1 = *(const v16h*)(kbase + (size_t)(j + 16 + n) * HDIM + K0);
        v8f s0 = {}, s1 = {};
        s0 = wmma_f16(aQ, bK0, s0);
        s1 = wmma_f16(aQ, bK1, s1);

        // ---- online softmax update (rows live across 16-lane halves)
#pragma unroll
        for (int i = 0; i < 8; ++i) {
            float a0 = rowOK[i] != 0.0f ? s0[i] * QSCALE : -10000.0f;
            float a1 = rowOK[i] != 0.0f ? s1[i] * QSCALE : -10000.0f;
            float mx = fmaxf(a0, a1);
            mx = fmaxf(mx, __shfl_xor(mx, 1, 32));
            mx = fmaxf(mx, __shfl_xor(mx, 2, 32));
            mx = fmaxf(mx, __shfl_xor(mx, 4, 32));
            mx = fmaxf(mx, __shfl_xor(mx, 8, 32));
            const float mNew = fmaxf(mI[i], mx);
            const float corr = __expf(mI[i] - mNew);
            const float p0 = __expf(a0 - mNew);
            const float p1 = __expf(a1 - mNew);
            float rs = p0 + p1;
            rs += __shfl_xor(rs, 1, 32);
            rs += __shfl_xor(rs, 2, 32);
            rs += __shfl_xor(rs, 4, 32);
            rs += __shfl_xor(rs, 8, 32);
            lI[i] = lI[i] * corr + rs;
            mI[i] = mNew;
            o0[i] *= corr;
            o1[i] *= corr;
            // stage P (C-layout -> row-major 16x32 f16 tile in LDS)
            ldsP[(rowb + i) * 32 + n]      = (_Float16)p0;
            ldsP[(rowb + i) * 32 + 16 + n] = (_Float16)p1;
        }
        __syncthreads();                       // 1-wave WG: waitcnt + nop
        const v16h aP = frag_a_from_h(ldsP, 32, lane);

        // ---- O += P @ V (vT rows -> contiguous B-frag loads)
        v16h bV0 = *(const v16h*)(vbase + (size_t)n * MKV + j + K0);
        v16h bV1 = *(const v16h*)(vbase + (size_t)(16 + n) * MKV + j + K0);
        o0 = wmma_f16(aP, bV0, o0);
        o1 = wmma_f16(aP, bV1, o1);
        __syncthreads();
    }

    // epilogue: normalize, write f16 into (B, N, 256) for the out-projection
#pragma unroll
    for (int i = 0; i < 8; ++i) {
        const float inv = 1.0f / lI[i];
        const int qrow = q0 + rowb + i;
        _Float16* dst = aoh + ((size_t)b * NQ + qrow) * DIMX + h * HDIM;
        dst[n]      = (_Float16)(o0[i] * inv);
        dst[16 + n] = (_Float16)(o1[i] * inv);
    }
}

// ---------------------------------------------------------------------------
// Output projection: out = attn_out @ Wp + bp   (fp32 out)
// grid = (B*N/16, 4), block = 32
// ---------------------------------------------------------------------------
__global__ void out_proj_kernel(const _Float16* __restrict__ aoh,
                                const _Float16* __restrict__ WpP,
                                const float* __restrict__ bp,
                                float* __restrict__ out) {
    const int lane = threadIdx.x;
    const int row0 = blockIdx.x * 16;     // row in B*N
    const int t0   = blockIdx.y * 4;
    v8f acc[4] = {};
    gemm16x64(aoh + (size_t)row0 * DIMX, WpP, t0, lane, acc);

    const int n   = lane & 15;
    const int mrb = (lane & 16) >> 1;
#pragma unroll
    for (int t = 0; t < 4; ++t) {
        const int col = (t0 + t) * 16 + n;
        const float bias = bp[col];
#pragma unroll
        for (int i = 0; i < 8; ++i)
            out[(size_t)(row0 + mrb + i) * DIMX + col] = acc[t][i] + bias;
    }
}

// ---------------------------------------------------------------------------
extern "C" void kernel_launch(void* const* d_in, const int* in_sizes, int n_in,
                              void* d_out, int out_size, void* d_ws, size_t ws_size,
                              hipStream_t stream) {
    const float*         x    = (const float*)d_in[0];        // (B,N,256)
    const float*         ctx  = (const float*)d_in[1];        // (B,M,256)
    const unsigned char* mask = (const unsigned char*)d_in[2];// (B,N) bool
    const float*         Wq   = (const float*)d_in[3];
    const float*         bq   = (const float*)d_in[4];
    const float*         Wkv  = (const float*)d_in[5];
    const float*         bkv  = (const float*)d_in[6];
    const float*         Wp   = (const float*)d_in[7];
    const float*         bp   = (const float*)d_in[8];
    float* out = (float*)d_out;

    // workspace layout (256B-aligned blocks)
    char* ws = (char*)d_ws;
    const size_t MB = 1024 * 1024;
    _Float16* xh   = (_Float16*)(ws);                    //  8 MiB  x in f16
    _Float16* ch   = (_Float16*)(ws +  8 * MB);          //  2 MiB  context f16
    _Float16* qf   = (_Float16*)(ws + 10 * MB);          //  8 MiB  Q heads
    _Float16* kf   = (_Float16*)(ws + 18 * MB);          //  2 MiB  K heads
    _Float16* vT   = (_Float16*)(ws + 20 * MB);          //  2 MiB  V^T heads
    _Float16* aoh  = (_Float16*)(ws + 22 * MB);          //  8 MiB  attn out f16
    _Float16* WqP  = (_Float16*)(ws + 30 * MB);          // 128 KiB packed Wq
    _Float16* WkvP = (_Float16*)(ws + 30 * MB + 256 * 1024);  // 256 KiB packed Wkv
    _Float16* WpP  = (_Float16*)(ws + 30 * MB + 768 * 1024);  // 128 KiB packed Wp

    const dim3 wave(32, 1, 1);   // one wave32 per WG (EXEC all-ones for WMMA)

    // one-time conversions / packs (tiny)
    cvt_f16_kernel<<<BATCH * NQ  * DIMX / (8 * 256), 256, 0, stream>>>(x,   xh);
    cvt_f16_kernel<<<BATCH * MKV * DIMX / (8 * 256), 256, 0, stream>>>(ctx, ch);
    pack_w_kernel<<<dim3(DIMX / 16,     KBLK), wave, 0, stream>>>(Wq,  WqP,  DIMX);
    pack_w_kernel<<<dim3(2 * DIMX / 16, KBLK), wave, 0, stream>>>(Wkv, WkvP, 2 * DIMX);
    pack_w_kernel<<<dim3(DIMX / 16,     KBLK), wave, 0, stream>>>(Wp,  WpP,  DIMX);

    // projections (WMMA GEMMs)
    q_proj_kernel  <<<dim3(BATCH * NQ / 16, 4),  wave, 0, stream>>>(xh, WqP, bq, qf);
    kv_proj_kernel <<<dim3(BATCH * MKV / 16, 8), wave, 0, stream>>>(ch, WkvP, bkv, kf, vT);

    // flash attention core
    flash_attn_kernel<<<dim3(BATCH * NHEADS, NQ / 16), wave, 0, stream>>>(qf, kf, vT, mask, aoh);

    // output projection
    out_proj_kernel<<<dim3(BATCH * NQ / 16, 4), wave, 0, stream>>>(aoh, WpP, bp, out);
}
